// multihead_self_attention_24781961298317
// MI455X (gfx1250) — matrix operand order repaired and checked by reading, not compiled, not run on hardware
//
#include <hip/hip_runtime.h>

// Problem constants (match reference)
#define DM   1024          // D_MODEL
#define NH   16            // heads
#define DH   64            // head dim
#define SQ   2048          // SEQ
#define BT   4             // BATCH
#define NTOK (BT * SQ)     // 8192 tokens
// ln(10000)/32  (RoPE: freq_j = exp(-j * ln(theta)/ (dh/2)))
#define ROPE_LN_OVER 0.28782313662425575f

typedef __attribute__((ext_vector_type(16))) _Float16 v16h;
typedef __attribute__((ext_vector_type(8)))  _Float16 v8h;
typedef __attribute__((ext_vector_type(4)))  _Float16 v4h;
typedef __attribute__((ext_vector_type(8)))  float    v8f;

// ---------------------------------------------------------------------------
// WMMA helpers (CDNA5 f32 = f16 x f16 + f32, 16x16x32)
// ---------------------------------------------------------------------------
__device__ __forceinline__ v8f wmma_f16(v16h a, v16h b, v8f c) {
  // (neg_a, A, neg_b, B, c_mod, C, reuse_a, reuse_b)
  return __builtin_amdgcn_wmma_f32_16x16x32_f16(false, a, false, b, (short)0, c,
                                                false, false);
}

// A-matrix 16x32 fragment (16-bit): lane L holds row M=L%16; element i has
// K = (L/16)*16 + i  ->  two contiguous 8-half (16B) chunks.
// rowptr must point at &src[row][k0].
__device__ __forceinline__ v16h frag_a16x32(const _Float16* rowptr, int lhalf) {
  const v8h lo = *(const v8h*)(rowptr + lhalf * 16);
  const v8h hi = *(const v8h*)(rowptr + lhalf * 16 + 8);
  v16h r;
#pragma unroll
  for (int i = 0; i < 8; ++i) { r[i] = lo[i]; r[i + 8] = hi[i]; }
  return r;
}

// B-matrix 32x16 fragment (16-bit): lane L holds column N=L%16, elements i
// cover K = (L/16)*16 + i  ->  16 contiguous halfs (32B) from the N-major row.
// p must point at &src[n][k0 + (L/16)*16].
__device__ __forceinline__ v16h frag_b32x16(const _Float16* p) {
  return *(const v16h*)p;
}

// ---------------------------------------------------------------------------
// f32 -> f16 convert (vectorized x4)
// ---------------------------------------------------------------------------
__global__ void cvt_f32_to_f16(const float4* __restrict__ src,
                               _Float16* __restrict__ dst, int n4) {
  int i = blockIdx.x * blockDim.x + threadIdx.x;
  if (i < n4) {
    float4 v = src[i];
    v4h o = { (_Float16)v.x, (_Float16)v.y, (_Float16)v.z, (_Float16)v.w };
    *(v4h*)(dst + (size_t)i * 4) = o;
  }
}

// ---------------------------------------------------------------------------
// QKV projection + fused RoPE (+1/sqrt(dh) folded into Q).
// grid = (NTOK/64, DM/32, 3) , block = 32 (one wave -> 32x64 output tile)
// Wh = [3][DM][DM] f16 (q,k,v).  Outputs:
//   Qh, Kh : [B,H,S,DH] f16 (roped; Q pre-scaled by 0.125)
//   Vt     : [B,H,DH,S] f16 (transposed so P·V B-fragments are contiguous)
// ---------------------------------------------------------------------------
__global__ void qkv_proj_rope(const _Float16* __restrict__ Xh,
                              const _Float16* __restrict__ Wh,
                              _Float16* __restrict__ Qh,
                              _Float16* __restrict__ Kh,
                              _Float16* __restrict__ Vt) {
  const int lane  = threadIdx.x & 31;
  const int lmod  = lane & 15;
  const int lhalf = lane >> 4;
  const int w  = blockIdx.z;          // 0=Q 1=K 2=V
  const int m0 = blockIdx.y * 32;     // output-feature block
  const int n0 = blockIdx.x * 64;     // token block
  const _Float16* Wsel = Wh + (size_t)w * DM * DM;

  const _Float16* arow0 = Wsel + (size_t)(m0 + lmod) * DM;
  const _Float16* arow1 = arow0 + (size_t)16 * DM;
  const _Float16* brow[4];
#pragma unroll
  for (int ni = 0; ni < 4; ++ni)
    brow[ni] = Xh + (size_t)(n0 + ni * 16 + lmod) * DM + lhalf * 16;

  v8f acc[2][4] = {};
  for (int k0 = 0; k0 < DM; k0 += 32) {
    if (k0 + 64 < DM) {
      __builtin_prefetch(arow0 + k0 + 64, 0, 1);
      __builtin_prefetch(brow[0] + k0 + 64, 0, 1);
    }
    v16h a0 = frag_a16x32(arow0 + k0, lhalf);
    v16h a1 = frag_a16x32(arow1 + k0, lhalf);
#pragma unroll
    for (int ni = 0; ni < 4; ++ni) {
      v16h b = frag_b32x16(brow[ni] + k0);
      acc[0][ni] = wmma_f16(a0, b, acc[0][ni]);
      acc[1][ni] = wmma_f16(a1, b, acc[1][ni]);
    }
  }

  // Epilogue: lane element r of tile (mi,ni) is
  //   token  n = n0 + ni*16 + lmod
  //   feature m = m0 + mi*16 + lhalf*8 + r
#pragma unroll
  for (int mi = 0; mi < 2; ++mi) {
    const int mbase = m0 + mi * 16 + lhalf * 8;   // even, pairs are (r, r+1)
#pragma unroll
    for (int ni = 0; ni < 4; ++ni) {
      const int n = n0 + ni * 16 + lmod;
      const int bidx = n >> 11;                   // n / SQ
      const int s = n & (SQ - 1);
      v8f c = acc[mi][ni];
      if (w < 2) {
        _Float16* dst = (w == 0) ? Qh : Kh;
#pragma unroll
        for (int t = 0; t < 4; ++t) {
          const int me = mbase + 2 * t;           // even feature index
          const int j = (me & (DH - 1)) >> 1;     // rope pair index 0..31
          const float freq = __expf(-(float)j * ROPE_LN_OVER);
          const float ang = (float)s * freq;
          float sn, cs;
          __sincosf(ang, &sn, &cs);
          float e = c[2 * t] * cs - c[2 * t + 1] * sn;
          float o = c[2 * t] * sn + c[2 * t + 1] * cs;
          if (w == 0) { e *= 0.125f; o *= 0.125f; }   // 1/sqrt(dh)
          const int h = me >> 6, d = me & (DH - 1);
          const size_t base = (((size_t)bidx * NH + h) * SQ + s) * DH + d;
          dst[base]     = (_Float16)e;
          dst[base + 1] = (_Float16)o;
        }
      } else {
#pragma unroll
        for (int r = 0; r < 8; ++r) {
          const int m = mbase + r;
          const int h = m >> 6, d = m & (DH - 1);
          Vt[(((size_t)bidx * NH + h) * DH + d) * SQ + s] = (_Float16)c[r];
        }
      }
    }
  }
}

// ---------------------------------------------------------------------------
// Flash attention with causal mask. grid = (SQ/16, NH, BT), block = 32.
// One wave owns 16 query rows; iterates 32-column key blocks.
// Output attnh: [B,S,D] f16 (token-major, ready as GEMM B-operand).
// ---------------------------------------------------------------------------
__global__ void flash_attn(const _Float16* __restrict__ Qh,
                           const _Float16* __restrict__ Kh,
                           const _Float16* __restrict__ Vt,
                           _Float16* __restrict__ attnh) {
  __shared__ _Float16 psh[16 * 32] __attribute__((aligned(64)));

  const int lane  = threadIdx.x & 31;
  const int lmod  = lane & 15;
  const int lhalf = lane >> 4;
  const int q0 = blockIdx.x * 16;
  const int h  = blockIdx.y;
  const int b  = blockIdx.z;

  const size_t head = ((size_t)b * NH + h);
  const _Float16* Qp = Qh + head * SQ * DH;
  const _Float16* Kp = Kh + head * SQ * DH;
  const _Float16* Vp = Vt + head * DH * SQ;   // [DH][SQ]

  // Q fragments (K=64 -> two 16x32 A-frags), Q pre-scaled by 1/8.
  const _Float16* qrow = Qp + (size_t)(q0 + lmod) * DH;
  const v16h qa0 = frag_a16x32(qrow, lhalf);
  const v16h qa1 = frag_a16x32(qrow + 32, lhalf);

  v8f o0 = {}, o1 = {}, o2 = {}, o3 = {};
  float mrow[8], lrow[8];
#pragma unroll
  for (int r = 0; r < 8; ++r) { mrow[r] = -__builtin_inff(); lrow[r] = 0.f; }

  const int nJ = (q0 + 47) >> 5;              // ceil((q0+16)/32)
  for (int jb = 0; jb < nJ; ++jb) {
    const int j0 = jb * 32;

    // ---- scores: S = Q · K^T  (two 16-col tiles, K-dim 64) ----
    v8f s0 = {}, s1 = {};
    {
      const _Float16* kr0 = Kp + (size_t)(j0 + lmod) * DH + lhalf * 16;
      s0 = wmma_f16(qa0, frag_b32x16(kr0), s0);
      s0 = wmma_f16(qa1, frag_b32x16(kr0 + 32), s0);
      const _Float16* kr1 = Kp + (size_t)(j0 + 16 + lmod) * DH + lhalf * 16;
      s1 = wmma_f16(qa0, frag_b32x16(kr1), s1);
      s1 = wmma_f16(qa1, frag_b32x16(kr1 + 32), s1);
    }

    // ---- causal mask + online softmax (rows live across 16-lane groups) ----
#pragma unroll
    for (int r = 0; r < 8; ++r) {
      const int qi = q0 + lhalf * 8 + r;      // query row of element r
      if (j0 + lmod > qi)      s0[r] = -__builtin_inff();
      if (j0 + 16 + lmod > qi) s1[r] = -__builtin_inff();

      float rm = fmaxf(s0[r], s1[r]);
#pragma unroll
      for (int m = 1; m < 16; m <<= 1) rm = fmaxf(rm, __shfl_xor(rm, m, 32));
      const float nm = fmaxf(mrow[r], rm);
      const float p0 = __expf(s0[r] - nm);
      const float p1 = __expf(s1[r] - nm);
      float rs = p0 + p1;
#pragma unroll
      for (int m = 1; m < 16; m <<= 1) rs += __shfl_xor(rs, m, 32);
      const float scale = __expf(mrow[r] - nm);
      lrow[r] = lrow[r] * scale + rs;
      mrow[r] = nm;
      o0[r] *= scale; o1[r] *= scale; o2[r] *= scale; o3[r] *= scale;

      const int M = lhalf * 8 + r;            // stage P (C-layout -> LDS)
      psh[M * 32 + lmod]      = (_Float16)p0;
      psh[M * 32 + 16 + lmod] = (_Float16)p1;
    }
    __syncthreads();

    // ---- O += P · V : reload P in A-layout, V^T rows are contiguous ----
    const v16h pa = frag_a16x32(&psh[lmod * 32], lhalf);
    {
      const size_t vko = (size_t)j0 + lhalf * 16;
      o0 = wmma_f16(pa, frag_b32x16(Vp + (size_t)(0 * 16 + lmod) * SQ + vko), o0);
      o1 = wmma_f16(pa, frag_b32x16(Vp + (size_t)(1 * 16 + lmod) * SQ + vko), o1);
      o2 = wmma_f16(pa, frag_b32x16(Vp + (size_t)(2 * 16 + lmod) * SQ + vko), o2);
      o3 = wmma_f16(pa, frag_b32x16(Vp + (size_t)(3 * 16 + lmod) * SQ + vko), o3);
    }
    __syncthreads();
  }

  // ---- normalize and emit token-major f16 [B,S,D] ----
#pragma unroll
  for (int r = 0; r < 8; ++r) {
    const float inv = 1.0f / lrow[r];
    const int s = q0 + lhalf * 8 + r;
    const size_t base = ((size_t)b * SQ + s) * DM + h * DH;
    attnh[base + 0 * 16 + lmod] = (_Float16)(o0[r] * inv);
    attnh[base + 1 * 16 + lmod] = (_Float16)(o1[r] * inv);
    attnh[base + 2 * 16 + lmod] = (_Float16)(o2[r] * inv);
    attnh[base + 3 * 16 + lmod] = (_Float16)(o3[r] * inv);
  }
}

// ---------------------------------------------------------------------------
// Output projection: out[n, m] = sum_k Wo[m,k] * attn[n,k], f32 result.
// grid = (NTOK/64, DM/32), block = 32.
// ---------------------------------------------------------------------------
__global__ void out_proj(const _Float16* __restrict__ Ah,
                         const _Float16* __restrict__ Woh,
                         float* __restrict__ out) {
  const int lane  = threadIdx.x & 31;
  const int lmod  = lane & 15;
  const int lhalf = lane >> 4;
  const int m0 = blockIdx.y * 32;
  const int n0 = blockIdx.x * 64;

  const _Float16* arow0 = Woh + (size_t)(m0 + lmod) * DM;
  const _Float16* arow1 = arow0 + (size_t)16 * DM;
  const _Float16* brow[4];
#pragma unroll
  for (int ni = 0; ni < 4; ++ni)
    brow[ni] = Ah + (size_t)(n0 + ni * 16 + lmod) * DM + lhalf * 16;

  v8f acc[2][4] = {};
  for (int k0 = 0; k0 < DM; k0 += 32) {
    if (k0 + 64 < DM) {
      __builtin_prefetch(arow0 + k0 + 64, 0, 1);
      __builtin_prefetch(brow[0] + k0 + 64, 0, 1);
    }
    v16h a0 = frag_a16x32(arow0 + k0, lhalf);
    v16h a1 = frag_a16x32(arow1 + k0, lhalf);
#pragma unroll
    for (int ni = 0; ni < 4; ++ni) {
      v16h b = frag_b32x16(brow[ni] + k0);
      acc[0][ni] = wmma_f16(a0, b, acc[0][ni]);
      acc[1][ni] = wmma_f16(a1, b, acc[1][ni]);
    }
  }

#pragma unroll
  for (int mi = 0; mi < 2; ++mi)
#pragma unroll
    for (int ni = 0; ni < 4; ++ni) {
      const int n = n0 + ni * 16 + lmod;
      const int mbase = m0 + mi * 16 + lhalf * 8;
      v8f c = acc[mi][ni];
#pragma unroll
      for (int r = 0; r < 8; ++r)
        out[(size_t)n * DM + mbase + r] = c[r];
    }
}

// ---------------------------------------------------------------------------
// Host launcher
// ---------------------------------------------------------------------------
extern "C" void kernel_launch(void* const* d_in, const int* in_sizes, int n_in,
                              void* d_out, int out_size, void* d_ws, size_t ws_size,
                              hipStream_t stream) {
  const float* X  = (const float*)d_in[0];
  const float* Wq = (const float*)d_in[1];
  const float* Wk = (const float*)d_in[2];
  const float* Wv = (const float*)d_in[3];
  const float* Wo = (const float*)d_in[4];
  float* out = (float*)d_out;

  char* ws = (char*)d_ws;
  size_t off = 0;
  _Float16* Xh = (_Float16*)(ws + off); off += (size_t)NTOK * DM * 2;     // 16 MB
  _Float16* Wh = (_Float16*)(ws + off); off += (size_t)4 * DM * DM * 2;   //  8 MB
  _Float16* Qh = (_Float16*)(ws + off); off += (size_t)NTOK * DM * 2;     // 16 MB
  _Float16* Kh = (_Float16*)(ws + off); off += (size_t)NTOK * DM * 2;     // 16 MB
  _Float16* Vt = (_Float16*)(ws + off); off += (size_t)NTOK * DM * 2;     // 16 MB
  _Float16* Ah = (_Float16*)(ws + off); off += (size_t)NTOK * DM * 2;     // 16 MB
  (void)ws_size; (void)in_sizes; (void)n_in; (void)out_size;

  // 1) converts
  {
    const int n4 = NTOK * DM / 4;
    cvt_f32_to_f16<<<(n4 + 255) / 256, 256, 0, stream>>>((const float4*)X, Xh, n4);
    const int w4 = DM * DM / 4;
    cvt_f32_to_f16<<<(w4 + 255) / 256, 256, 0, stream>>>((const float4*)Wq, Wh + 0 * (size_t)DM * DM, w4);
    cvt_f32_to_f16<<<(w4 + 255) / 256, 256, 0, stream>>>((const float4*)Wk, Wh + 1 * (size_t)DM * DM, w4);
    cvt_f32_to_f16<<<(w4 + 255) / 256, 256, 0, stream>>>((const float4*)Wv, Wh + 2 * (size_t)DM * DM, w4);
    cvt_f32_to_f16<<<(w4 + 255) / 256, 256, 0, stream>>>((const float4*)Wo, Wh + 3 * (size_t)DM * DM, w4);
  }
  // 2) fused QKV projection + RoPE (+Q scaling)
  qkv_proj_rope<<<dim3(NTOK / 64, DM / 32, 3), 32, 0, stream>>>(Xh, Wh, Qh, Kh, Vt);
  // 3) causal flash attention
  flash_attn<<<dim3(SQ / 16, NH, BT), 32, 0, stream>>>(Qh, Kh, Vt, Ah);
  // 4) output projection (f32 result)
  out_proj<<<dim3(NTOK / 64, DM / 32), 32, 0, stream>>>(Ah, Wh + 3 * (size_t)DM * DM, out);
}
